// GroupedQueryAttention_21423296873303
// MI455X (gfx1250) — compile-verified, matrix operand order checked by reading, not run on hardware
//
#include <hip/hip_runtime.h>
#include <hip/hip_bf16.h>

// ---------------------------------------------------------------------------
// GQA attention for MI455X (gfx1250): bf16 WMMA pipeline, fp32 accumulate.
// B=1, S=2048, D_IN=4096, D_OUT=4096, 32 heads, 8 KV heads, head_dim=128.
// ---------------------------------------------------------------------------

typedef __attribute__((ext_vector_type(16))) __bf16 v16bf;
typedef __attribute__((ext_vector_type(8)))  float  v8f;

#define S_LEN   2048
#define D_IN    4096
#define D_OUT   4096
#define N_HEADS 32
#define N_KV    8
#define HD      128
#define SM_SCALE 0.08838834764831845f  // 1/sqrt(128)

union FragU { v16bf v; uint4 q[2]; };

// Load a bf16 fragment in the CDNA5 16-bit A/B layout:
// lane half 0 holds K = [0..7] and [16..23]; half 1 holds [8..15] and [24..31].
__device__ __forceinline__ v16bf ldfrag(const __bf16* base, int ld, int row, int k0, int half) {
  const __bf16* p = base + (size_t)row * ld + k0 + half * 8;
  FragU u;
  u.q[0] = *reinterpret_cast<const uint4*>(p);        // 8 bf16 = 16B
  u.q[1] = *reinterpret_cast<const uint4*>(p + 16);   // next K chunk
  return u.v;
}

__device__ __forceinline__ v8f wmma_bf16(v16bf a, v16bf b, v8f c) {
  return __builtin_amdgcn_wmma_f32_16x16x32_bf16(false, a, false, b, (short)0, c, false, false);
}

// ---------------------------------------------------------------------------
// fp32 -> bf16 convert, 8 elements per thread (b128 in/out)
// ---------------------------------------------------------------------------
__global__ void cvt_f32_bf16(const float* __restrict__ in, __bf16* __restrict__ out, int n8) {
  int i = blockIdx.x * blockDim.x + threadIdx.x;
  if (i >= n8) return;
  const float4* p = reinterpret_cast<const float4*>(in) + (size_t)i * 2;
  float4 lo = p[0], hi = p[1];
  union { __bf16 b[8]; uint4 q; } u;
  u.b[0] = (__bf16)lo.x; u.b[1] = (__bf16)lo.y; u.b[2] = (__bf16)lo.z; u.b[3] = (__bf16)lo.w;
  u.b[4] = (__bf16)hi.x; u.b[5] = (__bf16)hi.y; u.b[6] = (__bf16)hi.z; u.b[7] = (__bf16)hi.w;
  reinterpret_cast<uint4*>(out)[i] = u.q;
}

// ---------------------------------------------------------------------------
// GEMM: C[M,N] = A[M,K] * B[N,K]^T   (A,B bf16 row-major; C = bf16 or f32)
// One wave computes a 32x64 tile: 2 A-frags x 4 B-frags = 8 WMMAs per k-step.
// ---------------------------------------------------------------------------
template <typename OutT>
__global__ __launch_bounds__(256) void gemm_bf16_wmma(
    const __bf16* __restrict__ A, const __bf16* __restrict__ B,
    OutT* __restrict__ C, int M, int N, int K) {
  int gw     = (blockIdx.x * 256 + threadIdx.x) >> 5;  // global wave id
  int lane   = threadIdx.x & 31;
  int tilesN = N >> 6;                                 // 64-wide N tiles
  int tm = gw / tilesN;                                // 32-row M tiles
  int tn = gw % tilesN;
  if (tm * 32 >= M) return;                            // wave-uniform guard
  int half = lane >> 4, r = lane & 15;

  v8f acc[2][4] = {};
  for (int k0 = 0; k0 < K; k0 += 32) {
    v16bf a0 = ldfrag(A, K, tm * 32 + r,      k0, half);
    v16bf a1 = ldfrag(A, K, tm * 32 + 16 + r, k0, half);
#pragma unroll
    for (int j = 0; j < 4; ++j) {
      v16bf b = ldfrag(B, K, tn * 64 + j * 16 + r, k0, half);
      acc[0][j] = wmma_bf16(a0, b, acc[0][j]);
      acc[1][j] = wmma_bf16(a1, b, acc[1][j]);
    }
    // keep L2->WGP ahead of the matrix pipe (global_prefetch_b8)
    if (k0 + 32 < K) {
      __builtin_prefetch(A + (size_t)(tm * 32 + r) * K + k0 + 32, 0, 3);
      __builtin_prefetch(B + (size_t)(tn * 64 + r) * K + k0 + 32, 0, 3);
      __builtin_prefetch(B + (size_t)(tn * 64 + 32 + r) * K + k0 + 32, 0, 3);
    }
  }
  // C/D layout: element v -> row base + v + 8*half, col tn*64 + j*16 + r
#pragma unroll
  for (int mi = 0; mi < 2; ++mi)
#pragma unroll
    for (int j = 0; j < 4; ++j)
#pragma unroll
      for (int v = 0; v < 8; ++v)
        C[(size_t)(tm * 32 + mi * 16 + v + 8 * half) * N + tn * 64 + j * 16 + r] =
            (OutT)acc[mi][j][v];
}

// ---------------------------------------------------------------------------
// RoPE: out = in*cos + rotate_half(in)*sin   layout [S][heads][128]
// ---------------------------------------------------------------------------
__global__ void rope_kernel(const __bf16* __restrict__ in, __bf16* __restrict__ out,
                            const float* __restrict__ cosT, const float* __restrict__ sinT,
                            int heads) {
  int idx = blockIdx.x * blockDim.x + threadIdx.x;
  int total = S_LEN * heads * HD;
  if (idx >= total) return;
  int d = idx & (HD - 1);
  int s = idx / (heads * HD);
  float v = (float)in[idx];
  float rot = (d < 64) ? -(float)in[idx + 64] : (float)in[idx - 64];
  out[idx] = (__bf16)(v * cosT[s * HD + d] + rot * sinT[s * HD + d]);
}

// ---------------------------------------------------------------------------
// V transpose: Vsrc[S][N_KV*HD] -> Vt[N_KV][HD][S]  (so P*V B-frags are row loads)
// ---------------------------------------------------------------------------
__global__ void transpose_v(const __bf16* __restrict__ Vsrc, __bf16* __restrict__ Vt) {
  int idx = blockIdx.x * blockDim.x + threadIdx.x;
  if (idx >= N_KV * HD * S_LEN) return;
  int s  = idx & (S_LEN - 1);
  int d  = (idx >> 11) & (HD - 1);
  int kv = idx >> 18;
  Vt[idx] = Vsrc[(size_t)s * (N_KV * HD) + kv * HD + d];
}

// ---------------------------------------------------------------------------
// Flash attention: one wave per (head, 16-query tile). Key blocks of 32.
// Q tile (16x128) hoisted into registers for the whole key loop.
// Scores via 8 WMMAs (K=128), softmax staged through per-wave LDS,
// P*V via 8 WMMAs into a 16x128 f32 accumulator.
// ---------------------------------------------------------------------------
#define WAVE_SMEM 3200  // 2048B scores f32 + 1024B P bf16 + 64B scale + pad

__global__ __launch_bounds__(256) void attn_kernel(
    const __bf16* __restrict__ Q,   // [S][N_HEADS*HD] (roped)
    const __bf16* __restrict__ Kr,  // [S][N_KV*HD]    (roped)
    const __bf16* __restrict__ Vt,  // [N_KV][HD][S]
    __bf16* __restrict__ Ctx) {     // [S][N_HEADS*HD]
  __shared__ __align__(128) char smem[8 * WAVE_SMEM];
  int wib  = threadIdx.x >> 5;
  int lane = threadIdx.x & 31;
  int gw   = blockIdx.x * 8 + wib;
  int h    = gw >> 7;                  // 128 query tiles per head
  int qt   = gw & 127;
  int kv   = h >> 2;                   // GROUP = 4
  int half = lane >> 4, r = lane & 15;

  float*  Sld = (float*)(smem + wib * WAVE_SMEM);           // [16][32] f32
  __bf16* Pld = (__bf16*)(smem + wib * WAVE_SMEM + 2048);   // [16][32] bf16
  float*  Scl = (float*)(smem + wib * WAVE_SMEM + 3072);    // [16] f32

  const __bf16* Qbase = Q + (size_t)(qt * 16) * (N_HEADS * HD) + h * HD;

  // Q tile is invariant over the key loop: keep all 4 K-chunks in registers.
  v16bf qa[4];
#pragma unroll
  for (int kc = 0; kc < 4; ++kc)
    qa[kc] = ldfrag(Qbase, N_HEADS * HD, r, kc * 32, half);

  float mrun = -1e30f, lrun = 0.f;
  v8f acc[8] = {};
  int qidx = qt * 16 + r;
  int nkb  = (qt * 16 + 15) / 32 + 1;

  for (int kb = 0; kb < nkb; ++kb) {
    // ---- scores: Q(16x128) . K_block(32x128)^T -> two 16x16 f32 frags ----
    const __bf16* Kbase = Kr + (size_t)(kb * 32) * (N_KV * HD) + kv * HD;
    v8f sc0 = {}, sc1 = {};
#pragma unroll
    for (int kc = 0; kc < 4; ++kc) {
      v16bf b0 = ldfrag(Kbase, N_KV * HD, r,      kc * 32, half);
      v16bf b1 = ldfrag(Kbase, N_KV * HD, 16 + r, kc * 32, half);
      sc0 = wmma_bf16(qa[kc], b0, sc0);
      sc1 = wmma_bf16(qa[kc], b1, sc1);
    }
    if (kb + 1 < nkb) {  // prefetch next key block rows
      __builtin_prefetch(Kbase + (size_t)(32 + r) * (N_KV * HD), 0, 3);
      __builtin_prefetch(Kbase + (size_t)(48 + r) * (N_KV * HD), 0, 3);
    }
    // stage scores to LDS (C layout: row m = v+8*half, col n = r / 16+r)
#pragma unroll
    for (int v = 0; v < 8; ++v) {
      Sld[(v + 8 * half) * 32 + r]      = sc0[v];
      Sld[(v + 8 * half) * 32 + 16 + r] = sc1[v];
    }
    asm volatile("s_wait_dscnt 0" ::: "memory");

    // ---- online softmax: each lane owns row r (both halves redundant) ----
    float bm = -1e30f;
#pragma unroll
    for (int j = 0; j < 32; ++j) {
      float s = Sld[r * 32 + j] * SM_SCALE;
      s = (kb * 32 + j <= qidx) ? s : -1e30f;
      bm = fmaxf(bm, s);
    }
    float mnew  = fmaxf(mrun, bm);
    float alpha = __expf(mrun - mnew);
    float ps = 0.f;
#pragma unroll
    for (int j = 0; j < 32; ++j) {
      float s = Sld[r * 32 + j] * SM_SCALE;
      float p = (kb * 32 + j <= qidx) ? __expf(s - mnew) : 0.f;
      ps += p;
      Pld[r * 32 + j] = (__bf16)p;
    }
    lrun = lrun * alpha + ps;
    mrun = mnew;
    Scl[r] = alpha;
    asm volatile("s_wait_dscnt 0" ::: "memory");

    // rescale accumulator rows by alpha(m)
    float sa[8];
#pragma unroll
    for (int v = 0; v < 8; ++v) sa[v] = Scl[v + 8 * half];
#pragma unroll
    for (int dt = 0; dt < 8; ++dt)
#pragma unroll
      for (int v = 0; v < 8; ++v) acc[dt][v] *= sa[v];

    // ---- ctx += P(16x32) . V_block(32x128) ----
    v16bf pA = ldfrag(Pld, 32, r, 0, half);   // from LDS, A-frag layout
#pragma unroll
    for (int dt = 0; dt < 8; ++dt) {
      const __bf16* Vb = Vt + ((size_t)kv * HD + dt * 16) * S_LEN + kb * 32;
      v16bf b = ldfrag(Vb, S_LEN, r, 0, half);  // row n = dim, K = key index
      acc[dt] = wmma_bf16(pA, b, acc[dt]);
    }
  }

  // ---- normalize by row sums and store ctx bf16 [S][N_HEADS*HD] ----
  Scl[r] = 1.0f / lrun;
  asm volatile("s_wait_dscnt 0" ::: "memory");
  float inv[8];
#pragma unroll
  for (int v = 0; v < 8; ++v) inv[v] = Scl[v + 8 * half];
#pragma unroll
  for (int dt = 0; dt < 8; ++dt)
#pragma unroll
    for (int v = 0; v < 8; ++v)
      Ctx[(size_t)(qt * 16 + v + 8 * half) * (N_HEADS * HD) + h * HD + dt * 16 + r] =
          (__bf16)(acc[dt][v] * inv[v]);
}

// ---------------------------------------------------------------------------
// Host-side launch
// ---------------------------------------------------------------------------
extern "C" void kernel_launch(void* const* d_in, const int* in_sizes, int n_in,
                              void* d_out, int out_size, void* d_ws, size_t ws_size,
                              hipStream_t stream) {
  const float* x    = (const float*)d_in[0];
  const float* cosT = (const float*)d_in[1];
  const float* sinT = (const float*)d_in[2];
  const float* Wq   = (const float*)d_in[3];
  const float* Wk   = (const float*)d_in[4];
  const float* Wv   = (const float*)d_in[5];
  const float* Wo   = (const float*)d_in[6];
  float* out = (float*)d_out;
  char* ws = (char*)d_ws;

  // workspace layout (bytes)
  __bf16* X16  = (__bf16*)(ws + 0);           // 16 MB  [S][D_IN]
  __bf16* WQ16 = (__bf16*)(ws + 16777216);    // 32 MB  [D_OUT][D_IN]
  __bf16* WK16 = (__bf16*)(ws + 50331648);    //  8 MB  [1024][D_IN]
  __bf16* WV16 = (__bf16*)(ws + 58720256);    //  8 MB
  __bf16* WO16 = (__bf16*)(ws + 67108864);    // 32 MB  [D_OUT][D_OUT]
  __bf16* Q16  = (__bf16*)(ws + 100663296);   // 16 MB  [S][4096]
  __bf16* K16  = (__bf16*)(ws + 117440512);   //  4 MB  [S][1024]
  __bf16* V16  = (__bf16*)(ws + 121634816);   //  4 MB  [S][1024]
  __bf16* QR   = (__bf16*)(ws + 125829120);   // 16 MB  roped Q
  __bf16* KR   = (__bf16*)(ws + 142606336);   //  4 MB  roped K
  __bf16* VT   = (__bf16*)(ws + 146800640);   //  4 MB  [8][128][S]
  __bf16* CTX  = (__bf16*)(ws + 150994944);   // 16 MB  [S][4096]

  auto cvt = [&](const float* src, __bf16* dst, int n) {
    int n8 = n / 8;
    cvt_f32_bf16<<<(n8 + 255) / 256, 256, 0, stream>>>(src, dst, n8);
  };
  cvt(x,  X16,  S_LEN * D_IN);
  cvt(Wq, WQ16, D_OUT * D_IN);
  cvt(Wk, WK16, N_KV * HD * D_IN);
  cvt(Wv, WV16, N_KV * HD * D_IN);
  cvt(Wo, WO16, D_OUT * D_OUT);

  // QKV projections (waves = (M/32)*(N/64); 8 waves per 256-thread block)
  gemm_bf16_wmma<__bf16><<<(64 * 64) / 8, 256, 0, stream>>>(X16, WQ16, Q16, S_LEN, 4096, D_IN);
  gemm_bf16_wmma<__bf16><<<(64 * 16) / 8, 256, 0, stream>>>(X16, WK16, K16, S_LEN, 1024, D_IN);
  gemm_bf16_wmma<__bf16><<<(64 * 16) / 8, 256, 0, stream>>>(X16, WV16, V16, S_LEN, 1024, D_IN);

  // RoPE + V transpose
  rope_kernel<<<(S_LEN * N_HEADS * HD + 255) / 256, 256, 0, stream>>>(Q16, QR, cosT, sinT, N_HEADS);
  rope_kernel<<<(S_LEN * N_KV * HD + 255) / 256, 256, 0, stream>>>(K16, KR, cosT, sinT, N_KV);
  transpose_v<<<(N_KV * HD * S_LEN + 255) / 256, 256, 0, stream>>>(V16, VT);

  // flash attention: 32 heads * 128 query tiles = 4096 waves, 8 per block
  attn_kernel<<<4096 / 8, 256, 0, stream>>>(QR, KR, VT, CTX);

  // output projection -> fp32 d_out
  gemm_bf16_wmma<float><<<(64 * 64) / 8, 256, 0, stream>>>(CTX, WO16, out, S_LEN, D_OUT, D_OUT);
}
